// GQA_SWA_Flash_66924180407280
// MI455X (gfx1250) — matrix operand order repaired and checked by reading, not compiled, not run on hardware
//
#include <hip/hip_runtime.h>
#include <hip/hip_bf16.h>
#include <stdint.h>

// ---------------- problem constants ----------------
#define T_LEN   4096
#define E_DIM   2048
#define M_DIM   2048
#define DKV     512
#define N_HEADS 16
#define HD      128
#define SWA     1024

typedef __attribute__((ext_vector_type(16))) __bf16 v16bf;
typedef __attribute__((ext_vector_type(8)))  float  v8f;

union Frag { v16bf v; uint4 q[2]; };
union Acc  { v8f v; float f[8]; };

static __device__ __forceinline__ v8f wmma_bf16(v16bf a, v16bf b, v8f c) {
  // D = A(16x32 bf16) * B(32x16 bf16) + C(16x16 f32)
  return __builtin_amdgcn_wmma_f32_16x16x32_bf16(
      /*neg_a=*/false, a, /*neg_b=*/false, b,
      /*c_mod=*/(short)0, c, /*reuse_a=*/false, /*reuse_b=*/false);
}

static __device__ __forceinline__ unsigned short f2bf(float f) {
  union { float f; unsigned int u; } x; x.f = f;
  unsigned int u = x.u;
  unsigned int r = (u + 0x7FFFu + ((u >> 16) & 1u)) >> 16;
  return (unsigned short)r;
}

static __device__ __forceinline__ v8f zero8() {
  Acc z;
  #pragma unroll
  for (int e = 0; e < 8; ++e) z.f[e] = 0.f;
  return z.v;
}

// async global -> LDS copy, 16B per lane, tracked by ASYNCcnt
static __device__ __forceinline__ void async_copy_b128(unsigned short* lds_ptr,
                                                       const void* gptr) {
  unsigned lds_off = (unsigned)(uintptr_t)lds_ptr;  // low 32b of flat = LDS addr
  asm volatile("global_load_async_to_lds_b128 %0, %1, off"
               :: "v"(lds_off), "v"(gptr)
               : "memory");
}
static __device__ __forceinline__ void async_fence() {
  asm volatile("s_wait_asynccnt 0" ::: "memory");
}

// ---------------- f32 -> bf16 convert ----------------
__global__ void cvt_f32_bf16(const float* __restrict__ in,
                             unsigned short* __restrict__ out, int n) {
  int i = blockIdx.x * blockDim.x + threadIdx.x;
  int stride = gridDim.x * blockDim.x;
  for (; i < n; i += stride) out[i] = f2bf(in[i]);
}

// ---------------- tiled WMMA GEMM (double-buffered, async A staging) -------
// C[M,N] = A[M,K](bf16 rowmajor) * B[K,N](bf16 rowmajor)   f32 accumulate
// OUTF==1: float out + bias;  OUTF==0: bf16 out (no bias)
template <int OUTF>
__global__ __launch_bounds__(256)
void gemm_wmma(const unsigned short* __restrict__ A,
               const unsigned short* __restrict__ B,
               void* __restrict__ Cp,
               const float* __restrict__ bias,
               int M, int N, int K) {
  constexpr int BM = 128, BN = 128, KC = 32;
  __shared__ __align__(16) unsigned short lA[2][BM * KC];   // [m][k]  8KB ea
  __shared__ __align__(16) unsigned short lBt[2][BN * KC];  // [n][k]  8KB ea

  const int tid = threadIdx.x;
  const int lane = tid & 31;
  const int w  = tid >> 5;   // 8 waves
  const int wr = w & 3;      // 4 waves along M (32 rows each)
  const int wc = w >> 2;     // 2 waves along N (64 cols each)
  const int m0 = blockIdx.y * BM;
  const int n0 = blockIdx.x * BN;
  const int l16 = lane & 15;
  const int hi  = lane >> 4;
  const int kbA = hi ? 8 : 0;    // A-fragment K base for this half-wave
  const int kbB = hi ? 8 : 0;    // B-fragment K base for this half-wave

  v8f acc[2][4];
  #pragma unroll
  for (int i = 0; i < 2; ++i)
    #pragma unroll
    for (int j = 0; j < 4; ++j) acc[i][j] = zero8();

  auto stage = [&](int k0, int buf) {
    // A tile 128x32: pure copy -> async global->LDS, 512 x b128, 2 per thread
    #pragma unroll
    for (int it = 0; it < 2; ++it) {
      int idx = tid + it * 256;          // 0..511
      int r  = idx >> 2;                 // 0..127
      int c8 = (idx & 3) << 3;           // 0,8,16,24
      async_copy_b128(&lA[buf][r * KC + c8],
                      &A[(size_t)(m0 + r) * K + k0 + c8]);
    }
    // B tile 32x128 staged transposed: 4096 scalars, 16 per thread
    #pragma unroll
    for (int it = 0; it < 16; ++it) {
      int idx = tid + it * 256;          // 0..4095
      int kk = idx >> 7;                 // 0..31
      int nn = idx & 127;                // 0..127
      lBt[buf][nn * KC + kk] = B[(size_t)(k0 + kk) * N + n0 + nn];
    }
  };

  stage(0, 0);
  async_fence();
  __syncthreads();

  int buf = 0;
  for (int k0 = 0; k0 < K; k0 += KC) {
    if (k0 + KC < K) stage(k0 + KC, buf ^ 1);   // overlap next-tile staging

    Frag a[2], b[4];
    #pragma unroll
    for (int i = 0; i < 2; ++i) {
      int row = wr * 32 + i * 16 + l16;
      a[i].q[0] = *(const uint4*)&lA[buf][row * KC + kbA];
      a[i].q[1] = *(const uint4*)&lA[buf][row * KC + 16 + kbA];
    }
    #pragma unroll
    for (int j = 0; j < 4; ++j) {
      int col = wc * 64 + j * 16 + l16;
      b[j].q[0] = *(const uint4*)&lBt[buf][col * KC + kbB];
      b[j].q[1] = *(const uint4*)&lBt[buf][col * KC + kbB + 16];
    }
    #pragma unroll
    for (int i = 0; i < 2; ++i)
      #pragma unroll
      for (int j = 0; j < 4; ++j)
        acc[i][j] = wmma_bf16(a[i].v, b[j].v, acc[i][j]);

    async_fence();
    __syncthreads();
    buf ^= 1;
  }

  // epilogue: C/D layout -> element e is row (e + 8*hi), col = lane%16
  #pragma unroll
  for (int i = 0; i < 2; ++i) {
    #pragma unroll
    for (int j = 0; j < 4; ++j) {
      Acc t; t.v = acc[i][j];
      int colg = n0 + wc * 64 + j * 16 + l16;
      #pragma unroll
      for (int e = 0; e < 8; ++e) {
        int rowg = m0 + wr * 32 + i * 16 + e + hi * 8;
        if (OUTF) {
          ((float*)Cp)[(size_t)rowg * N + colg] = t.f[e] + bias[colg];
        } else {
          ((unsigned short*)Cp)[(size_t)rowg * N + colg] = f2bf(t.f[e]);
        }
      }
    }
  }
}

// ---------------- flash attention, GQA + sliding window ----------------
// block = 128 queries x 1 head; 8 waves, 16 query rows each.
__global__ __launch_bounds__(256)
void attn_swa(const unsigned short* __restrict__ Q,   // (T, 2048) bf16
              const unsigned short* __restrict__ Kp,  // (T, 512) bf16
              const unsigned short* __restrict__ V,   // (T, 512) bf16
              unsigned short* __restrict__ Out) {     // (T, 2048) bf16
  constexpr float SCALE = 0.08838834764831845f;  // 1/sqrt(128)
  __shared__ __align__(16) unsigned short lVt[HD * 32];     // [d][key]  8KB
  __shared__ __align__(16) unsigned short lP[8 * 16 * 32];  // per-wave P 8KB

  const int tid = threadIdx.x;
  const int lane = tid & 31;
  const int w = tid >> 5;
  const int l16 = lane & 15;
  const int hi = lane >> 4;
  const int h = blockIdx.y;           // head
  const int g = h >> 2;               // kv group (QG=4)
  const int qb = blockIdx.x * 128;
  const int qw0 = qb + w * 16;        // first query row of this wave

  const int qcol = h * HD;            // column base in Q / Out
  const int kcol = g * HD;            // column base in K / V

  const int kbA = hi ? 8 : 0;
  const int kbB = hi ? 8 : 0;

  // resident Q fragments: 16 rows x 128 d = 4 A-fragments
  Frag qf[4];
  {
    const unsigned short* qrow = Q + (size_t)(qw0 + l16) * M_DIM + qcol;
    #pragma unroll
    for (int ds = 0; ds < 4; ++ds) {
      qf[ds].q[0] = *(const uint4*)&qrow[ds * 32 + kbA];
      qf[ds].q[1] = *(const uint4*)&qrow[ds * 32 + 16 + kbA];
    }
  }

  Acc o[8];
  float m[8], l[8];
  #pragma unroll
  for (int e = 0; e < 8; ++e) { m[e] = -1e30f; l[e] = 0.f; }
  #pragma unroll
  for (int dt = 0; dt < 8; ++dt) o[dt].v = zero8();

  int s_begin = qb - (SWA - 1);
  if (s_begin < 0) s_begin = 0;
  s_begin &= ~31;
  const int s_end = qb + 127;

  for (int s0 = s_begin; s0 <= s_end; s0 += 32) {
    // stage V chunk transposed: 32 keys x 128 d -> lVt[d][key]
    #pragma unroll
    for (int it = 0; it < 16; ++it) {
      int idx = tid + it * 256;        // 0..4095
      int key = idx >> 7;
      int d   = idx & 127;
      lVt[d * 32 + key] = V[(size_t)(s0 + key) * DKV + kcol + d];
    }
    __syncthreads();

    // wave-uniform relevance predicate (EXEC stays all-1 inside)
    bool active = (s0 <= qw0 + 15) && (s0 + 31 >= qw0 - (SWA - 1));
    if (active) {
      // S = Q * K^T  (two 16x16 key tiles, K split over d in 4 WMMAs each)
      Acc s[2];
      #pragma unroll
      for (int t = 0; t < 2; ++t) {
        v8f sv = zero8();
        int key = s0 + t * 16 + l16;
        const unsigned short* krow = Kp + (size_t)key * DKV + kcol;
        #pragma unroll
        for (int ds = 0; ds < 4; ++ds) {
          Frag bf;
          bf.q[0] = *(const uint4*)&krow[ds * 32 + kbB];
          bf.q[1] = *(const uint4*)&krow[ds * 32 + kbB + 16];
          sv = wmma_bf16(qf[ds].v, bf.v, sv);
        }
        s[t].v = sv;
      }

      // online softmax over this chunk (rows live at element e, half hi)
      float corr[8];
      #pragma unroll
      for (int e = 0; e < 8; ++e) {
        int q = qw0 + e + hi * 8;
        float mx = -1e30f;
        #pragma unroll
        for (int t = 0; t < 2; ++t) {
          int key = s0 + t * 16 + l16;
          float sv = s[t].f[e] * SCALE;
          bool ok = (key <= q) && ((q - key) < SWA);
          sv = ok ? sv : -1e30f;
          s[t].f[e] = sv;
          mx = fmaxf(mx, sv);
        }
        #pragma unroll
        for (int off = 1; off < 16; off <<= 1)
          mx = fmaxf(mx, __shfl_xor(mx, off, 16));
        float mn = fmaxf(m[e], mx);
        corr[e] = __expf(m[e] - mn);
        float rs = 0.f;
        #pragma unroll
        for (int t = 0; t < 2; ++t) {
          float p = __expf(s[t].f[e] - mn);
          s[t].f[e] = p;
          rs += p;
        }
        #pragma unroll
        for (int off = 1; off < 16; off <<= 1)
          rs += __shfl_xor(rs, off, 16);
        l[e] = l[e] * corr[e] + rs;
        m[e] = mn;
      }
      #pragma unroll
      for (int dt = 0; dt < 8; ++dt)
        #pragma unroll
        for (int e = 0; e < 8; ++e) o[dt].f[e] *= corr[e];

      // P: C-layout -> LDS -> A-layout (per-wave region, same-wave RAW)
      unsigned short* pw = &lP[w * 512];
      #pragma unroll
      for (int t = 0; t < 2; ++t)
        #pragma unroll
        for (int e = 0; e < 8; ++e) {
          int rl = e + hi * 8;
          pw[rl * 32 + t * 16 + l16] = f2bf(s[t].f[e]);
        }
      asm volatile("s_wait_dscnt 0" ::: "memory");
      Frag pa;
      pa.q[0] = *(const uint4*)&pw[l16 * 32 + kbA];
      pa.q[1] = *(const uint4*)&pw[l16 * 32 + 16 + kbA];

      // O += P * V  (8 d-tiles of 16 columns)
      #pragma unroll
      for (int dt = 0; dt < 8; ++dt) {
        Frag bv;
        int d = dt * 16 + l16;
        bv.q[0] = *(const uint4*)&lVt[d * 32 + kbB];
        bv.q[1] = *(const uint4*)&lVt[d * 32 + kbB + 16];
        o[dt].v = wmma_bf16(pa.v, bv.v, o[dt].v);
      }
    }
    __syncthreads();
  }

  // normalize and write att (bf16)
  float inv[8];
  #pragma unroll
  for (int e = 0; e < 8; ++e) inv[e] = 1.f / l[e];
  #pragma unroll
  for (int dt = 0; dt < 8; ++dt) {
    int col = qcol + dt * 16 + l16;
    #pragma unroll
    for (int e = 0; e < 8; ++e) {
      int row = qw0 + e + hi * 8;
      Out[(size_t)row * M_DIM + col] = f2bf(o[dt].f[e] * inv[e]);
    }
  }
}

// ---------------- host launch ----------------
extern "C" void kernel_launch(void* const* d_in, const int* in_sizes, int n_in,
                              void* d_out, int out_size, void* d_ws, size_t ws_size,
                              hipStream_t stream) {
  const float* x  = (const float*)d_in[0];
  const float* Wq = (const float*)d_in[1];
  const float* Wk = (const float*)d_in[2];
  const float* Wv = (const float*)d_in[3];
  const float* Wo = (const float*)d_in[4];
  const float* bo = (const float*)d_in[5];

  char* ws = (char*)d_ws;
  size_t off = 0;
  auto carve = [&](size_t bytes) {
    char* p = ws + off;
    off += (bytes + 255) & ~(size_t)255;
    return (unsigned short*)p;
  };
  unsigned short* xb  = carve((size_t)T_LEN * E_DIM * 2);
  unsigned short* wqb = carve((size_t)E_DIM * M_DIM * 2);
  unsigned short* wkb = carve((size_t)E_DIM * DKV * 2);
  unsigned short* wvb = carve((size_t)E_DIM * DKV * 2);
  unsigned short* wob = carve((size_t)M_DIM * M_DIM * 2);
  unsigned short* qb  = carve((size_t)T_LEN * M_DIM * 2);
  unsigned short* kb  = carve((size_t)T_LEN * DKV * 2);
  unsigned short* vb  = carve((size_t)T_LEN * DKV * 2);
  unsigned short* ab  = carve((size_t)T_LEN * M_DIM * 2);

  // 1. convert inputs to bf16
  cvt_f32_bf16<<<2048, 256, 0, stream>>>(x,  xb,  T_LEN * E_DIM);
  cvt_f32_bf16<<<2048, 256, 0, stream>>>(Wq, wqb, E_DIM * M_DIM);
  cvt_f32_bf16<<<1024, 256, 0, stream>>>(Wk, wkb, E_DIM * DKV);
  cvt_f32_bf16<<<1024, 256, 0, stream>>>(Wv, wvb, E_DIM * DKV);
  cvt_f32_bf16<<<2048, 256, 0, stream>>>(Wo, wob, M_DIM * M_DIM);

  // 2. projections (bf16 out)
  gemm_wmma<0><<<dim3(M_DIM / 128, T_LEN / 128), 256, 0, stream>>>(
      xb, wqb, qb, nullptr, T_LEN, M_DIM, E_DIM);
  gemm_wmma<0><<<dim3(DKV / 128, T_LEN / 128), 256, 0, stream>>>(
      xb, wkb, kb, nullptr, T_LEN, DKV, E_DIM);
  gemm_wmma<0><<<dim3(DKV / 128, T_LEN / 128), 256, 0, stream>>>(
      xb, wvb, vb, nullptr, T_LEN, DKV, E_DIM);

  // 3. flash attention (GQA + sliding window)
  attn_swa<<<dim3(T_LEN / 128, N_HEADS), 256, 0, stream>>>(qb, kb, vb, ab);

  // 4. output projection with bias (f32 out)
  gemm_wmma<1><<<dim3(M_DIM / 128, T_LEN / 128), 256, 0, stream>>>(
      ab, wob, d_out, bo, T_LEN, M_DIM, M_DIM);
}
